// AvgPoolingSelfAttention_83674552860986
// MI455X (gfx1250) — matrix operand order repaired and checked by reading, not compiled, not run on hardware
//
#include <hip/hip_runtime.h>
#include <stdint.h>

#define B_   2
#define T_   4096
#define D_   1024
#define H_   16
#define DH_  64
#define KP_  4
#define TP_  (T_ / KP_)   // 1024 pooled keys

typedef __attribute__((ext_vector_type(16))) __bf16 v16bf;
typedef __attribute__((ext_vector_type(8)))  float  v8f;

union BF16Frag {
    v16bf v;
    unsigned short u[16];
    uint4 q[2];
};

__device__ __forceinline__ unsigned short f2bf(float f) {
    unsigned int x = __float_as_uint(f);
    x += 0x7FFFu + ((x >> 16) & 1u);       // round-to-nearest-even
    return (unsigned short)(x >> 16);
}

// ---------------------------------------------------------------- prep kernels
__global__ __launch_bounds__(256) void cvt_bf16_kernel(const float* __restrict__ src,
                                                       unsigned short* __restrict__ dst,
                                                       int n) {
    int i = blockIdx.x * 256 + threadIdx.x;
    if (i < n) dst[i] = f2bf(src[i]);
}

__global__ __launch_bounds__(256) void pool_bf16_kernel(const float* __restrict__ hidden,
                                                        unsigned short* __restrict__ pooled) {
    int i = blockIdx.x * 256 + threadIdx.x;            // B*TP*D threads exactly
    int b  = i / (TP_ * D_);
    int r  = i - b * (TP_ * D_);
    int tt = r / D_;
    int d  = r - tt * D_;
    const float* p = hidden + ((size_t)b * T_ + (size_t)tt * KP_) * D_ + d;
    float s = (p[0] + p[D_] + p[2 * D_] + p[3 * D_]) * 0.25f;
    pooled[i] = f2bf(s);
}

__global__ __launch_bounds__(256) void mask_kernel(const int* __restrict__ m,
                                                   float* __restrict__ maskadd) {
    int i = blockIdx.x * 256 + threadIdx.x;            // B*TP threads exactly
    int b = i / TP_, tt = i - b * TP_;
    const int* p = m + b * T_ + tt * KP_;
    int any = p[0] | p[1] | p[2] | p[3];
    maskadd[i] = any ? -10000.0f : 0.0f;
}

// ------------------------------------------------------------- projection GEMM
// C[row, col] = sum_d A[row, d] * W[col, d] + bias[col]
// One wave owns a 16x64 output strip. bf16 WMMA, f32 accumulate.
// TRANSPOSED_OUT==0 : out is bf16 [B, H, L, DH]
// TRANSPOSED_OUT==1 : out is bf16 [B, H, DH, L]   (for V, so PV B-frags are contiguous)
template <int TRANSPOSED_OUT>
__global__ __launch_bounds__(256)
void gemm_qkv(const unsigned short* __restrict__ A,      // [B*L, D] bf16
              const unsigned short* __restrict__ W,      // [D, D]  bf16 (row=out col)
              const float* __restrict__ bias,            // [D]
              unsigned short* __restrict__ outp,
              int L) {
    const int lane   = threadIdx.x & 31;
    const int wave   = blockIdx.x * (blockDim.x >> 5) + (threadIdx.x >> 5);
    const int NSTRIP = D_ / 64;                          // 16
    const int mtile  = wave / NSTRIP;
    const int nstrip = wave - mtile * NSTRIP;
    const int mb     = mtile * 16;
    const int n16    = lane & 15;
    const int hi     = lane >> 4;

    v8f acc[4] = {};
    const unsigned short* arow = A + (size_t)(mb + n16) * D_;

#pragma unroll 1
    for (int kb = 0; kb < D_; kb += 32) {
        BF16Frag a;                                       // A frag: M=n16, K set by hi
        a.q[0] = *(const uint4*)(arow + kb + hi * 8);
        a.q[1] = *(const uint4*)(arow + kb + 16 + hi * 8);
#pragma unroll
        for (int nt = 0; nt < 4; ++nt) {
            const int nbase = nstrip * 64 + nt * 16;
            const unsigned short* wrow = W + (size_t)(nbase + n16) * D_ + kb + hi * 8;
            BF16Frag bfr;                                 // B frag: col=n16, K set by hi (same as A)
            bfr.q[0] = *(const uint4*)(wrow);
            bfr.q[1] = *(const uint4*)(wrow + 16);
            acc[nt] = __builtin_amdgcn_wmma_f32_16x16x32_bf16(
                false, a.v, false, bfr.v, (short)0, acc[nt], false, false);
        }
    }

#pragma unroll
    for (int nt = 0; nt < 4; ++nt) {
        const int nbase = nstrip * 64 + nt * 16;
        const int col   = nbase + n16;
        const float bv  = bias[col];
        const int h  = col >> 6;
        const int dh = col & 63;
        if (TRANSPOSED_OUT) {
            const int row0 = mb + 8 * hi;
            const int b    = row0 / L;
            const int pos0 = row0 - b * L;
            BF16Frag pk;
#pragma unroll
            for (int vg = 0; vg < 8; ++vg) pk.u[vg] = f2bf(acc[nt][vg] + bv);
            *(uint4*)(outp + ((size_t)(b * H_ + h) * DH_ + dh) * (size_t)L + pos0) = pk.q[0];
        } else {
#pragma unroll
            for (int vg = 0; vg < 8; ++vg) {
                const int row = mb + vg + 8 * hi;
                const int b   = row / L;
                const int pos = row - b * L;
                outp[((size_t)(b * H_ + h) * L + pos) * DH_ + dh] = f2bf(acc[nt][vg] + bv);
            }
        }
    }
}

// ------------------------------------------------------------ flash attention
// One wave per 16-query tile. Per 32-key step: 4 WMMAs for scores, online
// softmax (C-frag -> LDS -> row-major -> probs A-frag), 4 WMMAs for P*V.
__global__ __launch_bounds__(256)
void flash_attn(const unsigned short* __restrict__ q,    // [B,H,T,DH] bf16
                const unsigned short* __restrict__ k,    // [B,H,TP,DH] bf16
                const unsigned short* __restrict__ vT,   // [B,H,DH,TP] bf16
                const float* __restrict__ maskadd,       // [B,TP]
                float* __restrict__ outp) {              // [B,T,D] f32
    __shared__ float lds_s[8][16][36];                   // per-wave 16x32 score tile (+pad)
    __shared__ float lds_st[8][16];                      // per-wave row stats

    const int lane  = threadIdx.x & 31;
    const int wslot = threadIdx.x >> 5;
    const int wave  = blockIdx.x * 8 + wslot;
    const int QT    = T_ / 16;                           // 256 query tiles per (b,h)
    const int bh    = wave / QT;
    const int qt    = wave - bh * QT;
    const int b     = bh / H_;
    const int h     = bh - b * H_;
    const int qb    = qt * 16;
    const int n16   = lane & 15;
    const int hi    = lane >> 4;

    // Q A-fragments (dh slices 0..31 and 32..63), loaded once
    BF16Frag Aq[2];
    const unsigned short* qrow = q + ((size_t)(b * H_ + h) * T_ + qb + n16) * DH_;
#pragma unroll
    for (int s = 0; s < 2; ++s) {
        Aq[s].q[0] = *(const uint4*)(qrow + s * 32 + hi * 8);
        Aq[s].q[1] = *(const uint4*)(qrow + s * 32 + 16 + hi * 8);
    }

    float m_run = -3.0e30f;
    float l_run = 0.0f;
    v8f ctx[4] = {};
    float* sp = &lds_s[wslot][0][0];

#pragma unroll 1
    for (int kb = 0; kb < TP_; kb += 32) {
        // ---- scores: two 16x16 tiles, K=64 in two 32-steps
#pragma unroll
        for (int tile = 0; tile < 2; ++tile) {
            v8f c = {};
#pragma unroll
            for (int s = 0; s < 2; ++s) {
                const unsigned short* kr =
                    k + ((size_t)(b * H_ + h) * TP_ + kb + tile * 16 + n16) * DH_ + s * 32 + hi * 8;
                BF16Frag bk;
                bk.q[0] = *(const uint4*)(kr);
                bk.q[1] = *(const uint4*)(kr + 16);
                c = __builtin_amdgcn_wmma_f32_16x16x32_bf16(
                    false, Aq[s].v, false, bk.v, (short)0, c, false, false);
            }
            const float ma = maskadd[b * TP_ + kb + tile * 16 + n16];   // per-lane key
#pragma unroll
            for (int j = 0; j < 8; ++j)
                sp[(j + 8 * hi) * 36 + tile * 16 + n16] = c[j] * 0.125f + ma;
        }
        asm volatile("s_wait_dscnt 0" ::: "memory");

        // ---- row view: lane (n16,hi) -> row n16, cols {hi*8..} and {16+hi*8..}
        float rv[16];
        {
            const float* rp = sp + n16 * 36;
            float4 r0 = *(const float4*)(rp + hi * 8);
            float4 r1 = *(const float4*)(rp + hi * 8 + 4);
            float4 r2 = *(const float4*)(rp + 16 + hi * 8);
            float4 r3 = *(const float4*)(rp + 16 + hi * 8 + 4);
            rv[0]=r0.x; rv[1]=r0.y; rv[2]=r0.z; rv[3]=r0.w;
            rv[4]=r1.x; rv[5]=r1.y; rv[6]=r1.z; rv[7]=r1.w;
            rv[8]=r2.x; rv[9]=r2.y; rv[10]=r2.z; rv[11]=r2.w;
            rv[12]=r3.x; rv[13]=r3.y; rv[14]=r3.z; rv[15]=r3.w;
        }
        float lm = rv[0];
#pragma unroll
        for (int j = 1; j < 16; ++j) lm = fmaxf(lm, rv[j]);
        lm = fmaxf(lm, __shfl_xor(lm, 16, 32));
        const float mnew = fmaxf(m_run, lm);
        const float corr = __expf(m_run - mnew);

        BF16Frag P;                                       // probs A-frag (halves == rv order)
        float ls = 0.0f;
#pragma unroll
        for (int j = 0; j < 16; ++j) {
            const float p = __expf(rv[j] - mnew);
            ls += p;
            P.u[j] = f2bf(p);
        }
        ls += __shfl_xor(ls, 16, 32);
        l_run = l_run * corr + ls;
        m_run = mnew;

        if (hi == 0) lds_st[wslot][n16] = corr;           // row -> stats
        asm volatile("s_wait_dscnt 0" ::: "memory");
        float cf[8];
#pragma unroll
        for (int vg = 0; vg < 8; ++vg) cf[vg] = lds_st[wslot][vg + 8 * hi];

        // ---- ctx = ctx*corr + P(16x32) * V(32x64)
#pragma unroll
        for (int nt = 0; nt < 4; ++nt) {
            const unsigned short* vr =
                vT + ((size_t)(b * H_ + h) * DH_ + nt * 16 + n16) * TP_ + kb + hi * 8;
            BF16Frag bv;
            bv.q[0] = *(const uint4*)(vr);
            bv.q[1] = *(const uint4*)(vr + 16);
#pragma unroll
            for (int vg = 0; vg < 8; ++vg) ctx[nt][vg] *= cf[vg];
            ctx[nt] = __builtin_amdgcn_wmma_f32_16x16x32_bf16(
                false, P.v, false, bv.v, (short)0, ctx[nt], false, false);
        }
    }

    const float invl = 1.0f / l_run;
    if (hi == 0) lds_st[wslot][n16] = invl;
    asm volatile("s_wait_dscnt 0" ::: "memory");
    float iv[8];
#pragma unroll
    for (int vg = 0; vg < 8; ++vg) iv[vg] = lds_st[wslot][vg + 8 * hi];

#pragma unroll
    for (int nt = 0; nt < 4; ++nt)
#pragma unroll
        for (int vg = 0; vg < 8; ++vg)
            outp[((size_t)b * T_ + qb + vg + 8 * hi) * D_ + h * 64 + nt * 16 + n16] =
                ctx[nt][vg] * iv[vg];
}

// ----------------------------------------------------------------- launcher
extern "C" void kernel_launch(void* const* d_in, const int* in_sizes, int n_in,
                              void* d_out, int out_size, void* d_ws, size_t ws_size,
                              hipStream_t stream) {
    (void)in_sizes; (void)n_in; (void)out_size; (void)ws_size;
    const float* hidden = (const float*)d_in[0];
    const int*   amask  = (const int*)d_in[1];
    const float* Wq = (const float*)d_in[2];
    const float* bq = (const float*)d_in[3];
    const float* Wk = (const float*)d_in[4];
    const float* bk = (const float*)d_in[5];
    const float* Wv = (const float*)d_in[6];
    const float* bv = (const float*)d_in[7];
    float* outp = (float*)d_out;

    char* ws = (char*)d_ws;
    size_t off = 0;
    auto carve = [&](size_t bytes) {
        char* p = ws + off;
        off = (off + bytes + 255) & ~(size_t)255;
        return p;
    };
    const size_t N_HID  = (size_t)B_ * T_ * D_;      // 8388608
    const size_t N_POOL = (size_t)B_ * TP_ * D_;     // 2097152
    const size_t N_W    = (size_t)D_ * D_;           // 1048576

    unsigned short* hid_bf  = (unsigned short*)carve(N_HID * 2);
    unsigned short* pool_bf = (unsigned short*)carve(N_POOL * 2);
    unsigned short* wq_bf   = (unsigned short*)carve(N_W * 2);
    unsigned short* wk_bf   = (unsigned short*)carve(N_W * 2);
    unsigned short* wv_bf   = (unsigned short*)carve(N_W * 2);
    unsigned short* q_bf    = (unsigned short*)carve(N_HID * 2);
    unsigned short* k_bf    = (unsigned short*)carve(N_POOL * 2);
    unsigned short* vT_bf   = (unsigned short*)carve(N_POOL * 2);
    float*          madd    = (float*)carve((size_t)B_ * TP_ * 4);

    // prep
    cvt_bf16_kernel<<<(int)(N_HID / 256), 256, 0, stream>>>(hidden, hid_bf, (int)N_HID);
    cvt_bf16_kernel<<<(int)(N_W / 256), 256, 0, stream>>>(Wq, wq_bf, (int)N_W);
    cvt_bf16_kernel<<<(int)(N_W / 256), 256, 0, stream>>>(Wk, wk_bf, (int)N_W);
    cvt_bf16_kernel<<<(int)(N_W / 256), 256, 0, stream>>>(Wv, wv_bf, (int)N_W);
    pool_bf16_kernel<<<(int)(N_POOL / 256), 256, 0, stream>>>(hidden, pool_bf);
    mask_kernel<<<(B_ * TP_) / 256, 256, 0, stream>>>(amask, madd);

    // projections: waves = (M/16) * (D/64); 8 waves per 256-thread block
    gemm_qkv<0><<<(T_ * B_ / 16) * 16 / 8, 256, 0, stream>>>(hid_bf, wq_bf, bq, q_bf, T_);   // 1024 blocks
    gemm_qkv<0><<<(TP_ * B_ / 16) * 16 / 8, 256, 0, stream>>>(pool_bf, wk_bf, bk, k_bf, TP_); // 256 blocks
    gemm_qkv<1><<<(TP_ * B_ / 16) * 16 / 8, 256, 0, stream>>>(pool_bf, wv_bf, bv, vT_bf, TP_);

    // attention: B*H*(T/16) = 8192 waves -> 1024 blocks of 8 waves
    flash_attn<<<(B_ * H_ * (T_ / 16)) / 8, 256, 0, stream>>>(q_bf, k_bf, vT_bf, madd, outp);
}
